// SelfAttention_77644418777145
// MI455X (gfx1250) — matrix operand order repaired and checked by reading, not compiled, not run on hardware
//
#include <hip/hip_runtime.h>

// MI455X (gfx1250) fused local self-attention, v2.
// ~2.6 GFLOP dense (radius-6 mask -> only 13/32 key-chunks per query row),
// 8 MB HBM traffic -> memory/launch bound at 23.3 TB/s. Flash-attention
// fusion; all matrix math on v_wmma_f32_16x16x32_bf16 (K=32 == KEY_SIZE),
// fp32 softmax state. scores_xy folded in as a rank-2 seeding WMMA.
// Contraction relabeled by the A-layout involution so P/O repacks need no
// cross-lane shuffles and A operands load contiguous 32B per lane.

typedef __attribute__((ext_vector_type(16))) __bf16 v16bf;
typedef __attribute__((ext_vector_type(8)))  float  v8f;

#define NC    64    // channels
#define NPIX  1024  // 32*32
#define KS    32    // key size

// Involution: blocks of 8 arranged [0,2,1,3]. perm[K_A(e,hf)] == hf*16+e.
__device__ __forceinline__ int permute32(int p) {
    int blk = p >> 3;
    int nb  = ((blk & 1) << 1) | (blk >> 1);
    return (nb << 3) | (p & 7);
}

// ---------------------------------------------------------------------------
// Pass 1: per-pixel channel contractions (q,k,v,q_xy,k_xy) in fp32, rounded
// once to bf16. Q/K rows stored in natural dim order; V key-permuted within each
// 32-key chunk (A-operand side of the relabeled contraction). One thread per (batch, pixel).
// ---------------------------------------------------------------------------
__global__ __launch_bounds__(256) void proj_kernel(
    const float* __restrict__ x,
    const float* __restrict__ Wq, const float* __restrict__ Wk,
    const float* __restrict__ Wv,
    const float* __restrict__ Wqxy, const float* __restrict__ Wkxy,
    __bf16* __restrict__ Qbf, __bf16* __restrict__ Kbf,
    __bf16* __restrict__ Vbf,
    float* __restrict__ qxy, float* __restrict__ kxy)
{
    __shared__ float w[3 * 2048 + 256];
    for (int i = threadIdx.x; i < 2048; i += 256) {
        w[i]        = Wq[i];
        w[2048 + i] = Wk[i];
        w[4096 + i] = Wv[i];
    }
    if (threadIdx.x < 128) {
        w[6144 + threadIdx.x] = Wqxy[threadIdx.x];
        w[6272 + threadIdx.x] = Wkxy[threadIdx.x];
    }
    __syncthreads();

    const int tid = blockIdx.x * 256 + threadIdx.x;   // 0..16383
    const int b   = tid >> 10;
    const int n   = tid & (NPIX - 1);

    float aq[KS], ak[KS], av[KS];
    #pragma unroll
    for (int k = 0; k < KS; ++k) { aq[k] = 0.f; ak[k] = 0.f; av[k] = 0.f; }
    float aqx = 0.f, aqy = 0.f, akx = 0.f, aky = 0.f;

    const float* xb = x + (size_t)b * NC * NPIX + n;
    for (int c = 0; c < NC; ++c) {
        const float xc = xb[(size_t)c * NPIX];
        #pragma unroll
        for (int k = 0; k < KS; ++k) {
            aq[k] = fmaf(w[k * 64 + c],        xc, aq[k]);
            ak[k] = fmaf(w[2048 + k * 64 + c], xc, ak[k]);
            av[k] = fmaf(w[4096 + k * 64 + c], xc, av[k]);
        }
        aqx = fmaf(w[6144 + c], xc, aqx);
        aqy = fmaf(w[6208 + c], xc, aqy);
        akx = fmaf(w[6272 + c], xc, akx);
        aky = fmaf(w[6336 + c], xc, aky);
    }

    // Q: B-operand side -> natural dim order (matches K).
    __bf16* qrow = Qbf + (size_t)(b * NPIX + n) * KS;
    #pragma unroll
    for (int p = 0; p < KS; ++p) qrow[p] = (__bf16)aq[p];

    // K: A-operand side -> natural dim order (perm cancels).
    __bf16* krow = Kbf + (size_t)(b * NPIX + n) * KS;
    #pragma unroll
    for (int d = 0; d < KS; ++d) krow[d] = (__bf16)ak[d];

    // V: [b][d][key], key order permuted within each 32-key chunk.
    #pragma unroll
    for (int d = 0; d < KS; ++d)
        Vbf[(size_t)(b * KS + d) * NPIX + (n & ~31) + permute32(n & 31)] = (__bf16)av[d];

    qxy[(size_t)(b * 2 + 0) * NPIX + n] = aqx;
    qxy[(size_t)(b * 2 + 1) * NPIX + n] = aqy;
    kxy[(size_t)(b * 2 + 0) * NPIX + n] = akx;
    kxy[(size_t)(b * 2 + 1) * NPIX + n] = aky;
}

// Wout (64x32 f32) -> bf16, dim-permuted order (A-operand side).
__global__ void wout_kernel(const float* __restrict__ Wout,
                            __bf16* __restrict__ Wob)
{
    const int c = threadIdx.x;  // 0..63
    #pragma unroll
    for (int d = 0; d < KS; ++d)
        Wob[c * KS + permute32(d)] = (__bf16)Wout[c * KS + d];
}

// ---------------------------------------------------------------------------
// Pass 2: fused flash-attention + output projection.
// One wave per (batch, 16-query strip). Chunk = 32 keys = one grid row kx,
// so the radius-6 mask restricts kc to [h-6, h+6] (13 of 32 chunks).
// Per chunk: 2 xy-seed WMMAs + 2 QK WMMAs + 2 PV WMMAs, branch-free body.
// ---------------------------------------------------------------------------
__device__ __forceinline__ v8f wmma_bf16(v16bf a, v16bf b, v8f c) {
    return __builtin_amdgcn_wmma_f32_16x16x32_bf16(
        false, a, false, b, (short)0, c, false, false);
}

__global__ __launch_bounds__(128) void attn_kernel(
    const __bf16* __restrict__ Qbf, const __bf16* __restrict__ Kbf,
    const __bf16* __restrict__ Vbf,
    const float* __restrict__ qxy, const float* __restrict__ kxy,
    const __bf16* __restrict__ Wob,
    float* __restrict__ out)
{
    const int lane = threadIdx.x & 31;
    const int wid  = blockIdx.x * 4 + (threadIdx.x >> 5);
    const int b    = wid >> 6;
    const int mt   = wid & 63;
    const int h    = mt >> 1;          // query grid row
    const int w0   = (mt & 1) << 4;    // query grid col base
    const int col  = lane & 15;
    const int hf   = lane >> 4;        // lane half
    const int m    = mt * 16 + col;    // flattened query index (= h*32+w0+col)
    const int qy   = w0 + col;

    // Q as B operand: lane holds query column m, contraction elems hf*16+e.
    const v16bf qB = *(const v16bf*)(Qbf + (size_t)(b * NPIX + m) * KS + hf * 16);

    // q_xy as B operand: only contraction dims 0,1 non-zero (half-0 lanes).
    const float hfm = hf ? 0.f : 1.f;
    v16bf zbf;
    #pragma unroll
    for (int e = 0; e < 16; ++e) zbf[e] = (__bf16)0.f;
    v16bf qxyB = zbf;
    qxyB[0] = (__bf16)(qxy[(size_t)(b * 2 + 0) * NPIX + m] * hfm);
    qxyB[1] = (__bf16)(qxy[(size_t)(b * 2 + 1) * NPIX + m] * hfm);

    const float* kxp0 = kxy + (size_t)(b * 2 + 0) * NPIX;
    const float* kxp1 = kxy + (size_t)(b * 2 + 1) * NPIX;

    v8f O0, O1;
    #pragma unroll
    for (int i = 0; i < 8; ++i) { O0[i] = 0.f; O1[i] = 0.f; }
    float runM = -1e30f, runS = 0.f;

    const int kc0 = (h - 6 < 0)  ? 0  : h - 6;
    const int kc1 = (h + 6 > 31) ? 31 : h + 6;

    for (int kc = kc0; kc <= kc1; ++kc) {
        const size_t kb = (size_t)(b * NPIX + kc * 32);

        // K as A operand: rows = keys kc*32+col (tile0) / +16 (tile1).
        const v16bf kA0 = *(const v16bf*)(Kbf + (kb + col)      * KS + hf * 16);
        const v16bf kA1 = *(const v16bf*)(Kbf + (kb + 16 + col) * KS + hf * 16);

        // k_xy as A operand: dims 0,1 only (half-0 lanes), per key row.
        v16bf xyA0 = zbf, xyA1 = zbf;
        xyA0[0] = (__bf16)(kxp0[kc * 32 + col] * hfm);
        xyA0[1] = (__bf16)(kxp1[kc * 32 + col] * hfm);
        xyA1[0] = (__bf16)(kxp0[kc * 32 + 16 + col] * hfm);
        xyA1[1] = (__bf16)(kxp1[kc * 32 + 16 + col] * hfm);

        // Prefetch next chunk's K/V (global_prefetch_b8).
        const int kn = (kc < kc1) ? (kc + 1) : kc;
        __builtin_prefetch(Kbf + ((size_t)(b * NPIX + kn * 32) + col) * KS, 0, 1);
        __builtin_prefetch(Vbf + (size_t)(b * KS + col) * NPIX + kn * 32, 0, 1);

        // S = K*Q + Kxy*Qxy  (scores + scores_xy, fp32 accumulate).
        v8f S0, S1;
        #pragma unroll
        for (int i = 0; i < 8; ++i) { S0[i] = 0.f; S1[i] = 0.f; }
        S0 = wmma_bf16(xyA0, qxyB, S0);
        S1 = wmma_bf16(xyA1, qxyB, S1);
        S0 = wmma_bf16(kA0, qB, S0);
        S1 = wmma_bf16(kA1, qB, S1);

        // Mask (pure arithmetic -> v_cndmask) + online softmax.
        const int dx  = kc - h;
        const int dx2 = dx * dx;
        float s0v[8], s1v[8];
        float cmax = -1e30f;
        #pragma unroll
        for (int i = 0; i < 8; ++i) {
            const int ky0 = i + 8 * hf;      // D row of S0 element i
            const int ky1 = ky0 + 16;        // D row of S1 element i
            const int dy0 = ky0 - qy, dy1 = ky1 - qy;
            const float v0 = (dx2 + dy0 * dy0 > 36) ? -1e30f : S0[i];
            const float v1 = (dx2 + dy1 * dy1 > 36) ? -1e30f : S1[i];
            s0v[i] = v0; s1v[i] = v1;
            cmax = fmaxf(cmax, fmaxf(v0, v1));
        }
        cmax = fmaxf(cmax, __shfl_xor(cmax, 16, 32));   // combine lane halves
        const float newM  = fmaxf(runM, cmax);
        const float scale = __expf(runM - newM);        // 0 flushes -1e30 prefix
        runM = newM;

        float p0[8], p1[8], csum = 0.f;
        #pragma unroll
        for (int i = 0; i < 8; ++i) {
            p0[i] = __expf(s0v[i] - newM);
            p1[i] = __expf(s1v[i] - newM);
            csum += p0[i] + p1[i];
        }
        csum += __shfl_xor(csum, 16, 32);
        runS = runS * scale + csum;

        // P repack D->B layout: with the perm relabeling this is just
        // [tile0 | tile1] in-lane -- no cross-lane exchange needed.
        v16bf P;
        #pragma unroll
        for (int i = 0; i < 8; ++i) {
            P[i]     = (__bf16)p0[i];
            P[i + 8] = (__bf16)p1[i];
        }

        #pragma unroll
        for (int i = 0; i < 8; ++i) { O0[i] *= scale; O1[i] *= scale; }

        // V as A operand: rows = head dims (tile0 d=0-15, tile1 d=16-31),
        // contraction over the 32 chunk keys (key-permuted storage).
        const v16bf vA0 = *(const v16bf*)(Vbf + (size_t)(b * KS + col)      * NPIX + kc * 32 + hf * 16);
        const v16bf vA1 = *(const v16bf*)(Vbf + (size_t)(b * KS + 16 + col) * NPIX + kc * 32 + hf * 16);
        O0 = wmma_bf16(vA0, P, O0);
        O1 = wmma_bf16(vA1, P, O1);
    }

    // Normalize; O repack D->B layout is again just [O0 | O1] in-lane.
    const float inv = 1.f / runS;   // runS > 0: self-key always unmasked
    v16bf oB;
    #pragma unroll
    for (int i = 0; i < 8; ++i) {
        oB[i]     = (__bf16)(O0[i] * inv);
        oB[i + 8] = (__bf16)(O1[i] * inv);
    }

    // Output projection: y[c,m] = sum_d Wout[c,d]*O[d,m]; 4 tiles of 16 rows.
    #pragma unroll
    for (int t = 0; t < 4; ++t) {
        const v16bf wA = *(const v16bf*)(Wob + (t * 16 + col) * KS + hf * 16);
        v8f Y;
        #pragma unroll
        for (int i = 0; i < 8; ++i) Y[i] = 0.f;
        Y = wmma_bf16(wA, oB, Y);
        #pragma unroll
        for (int i = 0; i < 8; ++i) {
            const int c = t * 16 + i + 8 * hf;
            out[(size_t)(b * NC + c) * NPIX + m] = Y[i];
        }
    }
}

// ---------------------------------------------------------------------------
extern "C" void kernel_launch(void* const* d_in, const int* in_sizes, int n_in,
                              void* d_out, int out_size, void* d_ws, size_t ws_size,
                              hipStream_t stream)
{
    (void)in_sizes; (void)n_in; (void)out_size; (void)ws_size;
    const float* x    = (const float*)d_in[0];
    const float* Wq   = (const float*)d_in[1];
    const float* Wk   = (const float*)d_in[2];
    const float* Wv   = (const float*)d_in[3];
    const float* Wqxy = (const float*)d_in[4];
    const float* Wkxy = (const float*)d_in[5];
    const float* Wout = (const float*)d_in[6];

    // Workspace layout (1 MB / 128 KB aligned -> 32B-aligned WMMA loads)
    char*  ws  = (char*)d_ws;
    __bf16* Qbf = (__bf16*)(ws);                              // 1 MB
    __bf16* Kbf = (__bf16*)(ws + (1u << 20));                 // 1 MB
    __bf16* Vbf = (__bf16*)(ws + (2u << 20));                 // 1 MB
    float*  qxy = (float*) (ws + (3u << 20));                 // 128 KB
    float*  kxy = (float*) (ws + (3u << 20) + (128u << 10));  // 128 KB
    __bf16* Wob = (__bf16*)(ws + (3u << 20) + (256u << 10));  // 4 KB

    proj_kernel<<<64, 256, 0, stream>>>(x, Wq, Wk, Wv, Wqxy, Wkxy,
                                        Qbf, Kbf, Vbf, qxy, kxy);
    wout_kernel<<<1, 64, 0, stream>>>(Wout, Wob);
    attn_kernel<<<256, 128, 0, stream>>>(Qbf, Kbf, Vbf, qxy, kxy, Wob,
                                         (float*)d_out);
}
